// EMALayer_90984587199065
// MI455X (gfx1250) — compile-verified, matrix operand order checked by reading, not run on hardware
//
#include <hip/hip_runtime.h>
#include <hip/hip_bf16.h>
#include <stddef.h>
#include <stdint.h>

// ---------------------------------------------------------------------------
// EMA layer for MI455X (gfx1250, wave32):
//   z = x @ W^T + b           (bf16x3 split-precision WMMA GEMM, TDM-staged B)
//   out = fwdscan(z; g1) + bwdscan(z; g2) + beta * z   (chunked parallel scan)
// ---------------------------------------------------------------------------

typedef __attribute__((ext_vector_type(16))) __bf16 v16bf;
typedef __attribute__((ext_vector_type(8)))  float  v8f;
typedef __attribute__((ext_vector_type(4)))  unsigned int v4u;
typedef __attribute__((ext_vector_type(8)))  int    v8i;
typedef __attribute__((ext_vector_type(4)))  int    v4i;

#define L_DIM 4096
#define B_DIM 8
#define E_DIM 1024
#define Z_DIM 256
#define M_DIM (L_DIM * B_DIM)   // 32768

#if defined(__has_builtin)
#  if __has_builtin(__builtin_amdgcn_tensor_load_to_lds) && \
      __has_builtin(__builtin_amdgcn_s_wait_tensorcnt)
#    define USE_TDM 1
#  endif
#endif
#ifndef USE_TDM
#  define USE_TDM 0
#endif

// ---------------------------------------------------------------------------
// Kernel 0: split W (Z x E, f32) into bf16 hi/lo pair for split-precision GEMM
// ---------------------------------------------------------------------------
__global__ __launch_bounds__(256)
void ema_convW(const float* __restrict__ W,
               __bf16* __restrict__ Whi, __bf16* __restrict__ Wlo)
{
    const int i = (blockIdx.x * 256 + threadIdx.x) * 4;   // 256 blocks * 256 thr * 4
#pragma unroll
    for (int r = 0; r < 4; ++r) {
        const int idx = i + r;
        const float w = W[idx];
        const __bf16 h = (__bf16)w;
        Whi[idx] = h;
        Wlo[idx] = (__bf16)(w - (float)h);
    }
}

#if USE_TDM
// ---------------------------------------------------------------------------
// TDM descriptor: load a 256(N) x 32(K) bf16 tile of W (row pitch 1024 elems)
// into LDS, HW-padding each 64B row with 16B -> 80B pitch (bank-conflict-free).
// D# fields per CDNA5 ISA ch.8: group0 = {flags, lds_addr, global_addr, type},
// group1 = {pad ctrl, tensor dims, tile dims, strides}. This toolchain's
// builtin takes 6 args: (g0 v4u, g1 v8i, g2 v4i, g3 v4i, extra v8i, cpol).
// ---------------------------------------------------------------------------
__device__ __forceinline__ void tdm_load_B(const __bf16* gsrc, uint32_t ldsOff)
{
    const uint64_t ga = (uint64_t)(uintptr_t)gsrc;

    v4u g0;
    g0[0] = 1u;                                          // count=1 (valid), no gather
    g0[1] = ldsOff;                                      // lds_addr (byte offset)
    g0[2] = (uint32_t)ga;                                // global_addr[31:0]
    g0[3] = (uint32_t)((ga >> 32) & 0x01FFFFFFu)         // global_addr[56:32]
          | (2u << 30);                                  // type = 2 ("image")

    v8i g1;
    g1[0] = (int)((1u << 16)      // data_size = 1 -> 2 bytes
                | (1u << 20)      // pad_enable
                | (3u << 22)      // pad_interval: 16 DWORDs (64B) between pads
                | (3u << 25));    // pad_amount: 4 DWORDs (16B)
    g1[1] = (int)(1024u << 16);   // tensor_dim0 = 1024 (K), bits[79:48]
    g1[2] = (int)(256u  << 16);   // tensor_dim0 hi=0 | tensor_dim1 = 256 (N)
    g1[3] = (int)(32u   << 16);   // tensor_dim1 hi=0 | tile_dim0 = 32
    g1[4] = 256;                  // tile_dim1 = 256 rows, tile_dim2 = 0
    g1[5] = 1024;                 // tensor_dim0_stride = 1024 elements
    g1[6] = 0;
    g1[7] = 0;

    v4i g2 = {0, 0, 0, 0};        // dims 2/3 unused (2D tile)
    v4i g3 = {0, 0, 0, 0};
    v8i g4 = {0, 0, 0, 0, 0, 0, 0, 0};   // extra group (6-arg form): zeros
    __builtin_amdgcn_tensor_load_to_lds(g0, g1, g2, g3, g4, 0);
}
#endif

// ---------------------------------------------------------------------------
// Kernel 1: GEMM  z[m, n] = sum_k x[m, k] * W[n, k] + b[n]
//   block tile 64(M) x 256(N), K-chunk 32, 8 waves; wave tile 32(M) x 64(N)
//   3 WMMAs per tile (hi*hi + hi*lo + lo*hi) ~ f32 accuracy
//   B staged by TDM into double-buffered LDS (DMA of chunk k+1 overlaps
//   the 24 WMMAs of chunk k); A converted f32->bf16 hi/lo by the VALU.
//   Dynamic LDS layout (92160 B total):
//     Ah  @     0 (5120)   Al @  5120 (5120)
//     Bh  @ 10240 ([2][256][40] = 2 x 20480)
//     Bl  @ 51200 ([2][256][40] = 2 x 20480)
//   B is indexed as rows [cur*256 + n] off a SINGLE base pointer so the
//   address-space inference keeps every fragment read on the DS path.
// ---------------------------------------------------------------------------
#define BH_OFF 10240u
#define BL_OFF 51200u
#define BBUF_BYTES 20480u

__global__ __launch_bounds__(256)
void ema_gemm(const float*  __restrict__ x,
              const __bf16* __restrict__ Whi,
              const __bf16* __restrict__ Wlo,
              const float*  __restrict__ bias,
              float*        __restrict__ z)
{
    constexpr int Kdim = E_DIM;
    constexpr int Ndim = Z_DIM;
    constexpr int AP   = 40;      // padded LDS pitch in bf16 (80B)

    extern __shared__ __align__(16) char smem[];
    __bf16 (*Ah)[AP] = (__bf16(*)[AP])(smem);              //  5120 B
    __bf16 (*Al)[AP] = (__bf16(*)[AP])(smem + 5120);       //  5120 B
    __bf16 (*Bh)[AP] = (__bf16(*)[AP])(smem + BH_OFF);     // [2*256][AP]
    __bf16 (*Bl)[AP] = (__bf16(*)[AP])(smem + BL_OFF);     // [2*256][AP]

    const int tid   = threadIdx.x;
    const int lane  = tid & 31;
    const int wave  = tid >> 5;
    const int mBase = blockIdx.x * 64;

    const int wm = wave & 1;           // 32-row half of the block tile
    const int wn = wave >> 1;          // 64-col quarter
    const int n0 = wn * 64;

    v8f acc[2][4];
#pragma unroll
    for (int i = 0; i < 2; ++i)
#pragma unroll
        for (int j = 0; j < 4; ++j)
#pragma unroll
            for (int e = 0; e < 8; ++e) acc[i][j][e] = 0.0f;

    // cooperative A staging assignment
    const int ar = tid >> 2;                 // 0..63  (A row)
    const int ac = (tid & 3) * 8;            // 0,8,16,24 (A col segment)
    const float* aptr = x + (size_t)(mBase + ar) * Kdim + ac;
#if !USE_TDM
    const __bf16* bhp = Whi + (size_t)tid * Kdim;   // one W row per thread
    const __bf16* blp = Wlo + (size_t)tid * Kdim;
#endif

    // WMMA fragment lane mapping (ISA 16-bit A/B layout, wave32)
    const int fm = lane & 15;                // M (or N) index within tile
    const int kb = (lane >> 4) * 8;          // K base: lanes 0-15 -> 0, 16-31 -> 8

    int cur = 0;
#if USE_TDM
    if (wave == 0) {                         // prologue: DMA chunk 0 into buf 0
        tdm_load_B(Whi, BH_OFF);
        tdm_load_B(Wlo, BL_OFF);
    }
#endif

    for (int kc = 0; kc < Kdim; kc += 32) {
#if USE_TDM
        const bool haveNext = (kc + 32 < Kdim);
        if (wave == 0 && haveNext) {         // prefetch chunk k+1 into other buf
            const uint32_t nb = (uint32_t)(cur ^ 1) * BBUF_BYTES;
            tdm_load_B(Whi + kc + 32, BH_OFF + nb);
            tdm_load_B(Wlo + kc + 32, BL_OFF + nb);
        }
#else
        {   // sync B staging (fallback): pre-split bf16 W rows into LDS
            const uint4* s0 = (const uint4*)(bhp + kc);
            const uint4* s1 = (const uint4*)(blp + kc);
            uint4* d0 = (uint4*)&Bh[cur * 256 + tid][0];  // 80B pitch, 16B aligned
            uint4* d1 = (uint4*)&Bl[cur * 256 + tid][0];
#pragma unroll
            for (int i = 0; i < 4; ++i) { d0[i] = s0[i]; d1[i] = s1[i]; }
        }
#endif
        // ---- stage A: f32 -> bf16 hi/lo split into LDS -------------------
        {
            const float4 a0 = *(const float4*)(aptr + kc);
            const float4 a1 = *(const float4*)(aptr + kc + 4);
            float av[8] = {a0.x, a0.y, a0.z, a0.w, a1.x, a1.y, a1.z, a1.w};
#pragma unroll
            for (int i = 0; i < 8; ++i) {
                const __bf16 h = (__bf16)av[i];
                Ah[ar][ac + i] = h;
                Al[ar][ac + i] = (__bf16)(av[i] - (float)h);
            }
        }
        if (kc + 32 < Kdim)
            __builtin_prefetch(aptr + kc + 32, 0, 0);   // global_prefetch_b8

#if USE_TDM
        if (wave == 0) {
            // wait for CURRENT chunk's 2 descriptors only (TDM in-order per
            // wave): <=2 outstanding leaves the k+1 prefetch in flight.
            if (haveNext) __builtin_amdgcn_s_wait_tensorcnt(2);
            else          __builtin_amdgcn_s_wait_tensorcnt(0);
        }
#endif
        __syncthreads();

        // ---- load fragments (all DS path) -------------------------------
        const int bRow = cur * 256;
        v16bf ah[2], al[2], bh[4], bl[4];
#pragma unroll
        for (int mi = 0; mi < 2; ++mi) {
            const __bf16* rh = &Ah[wm * 32 + mi * 16 + fm][kb];
            const __bf16* rl = &Al[wm * 32 + mi * 16 + fm][kb];
            union { v16bf v; uint4 q[2]; } u;
            u.q[0] = *(const uint4*)rh; u.q[1] = *(const uint4*)(rh + 16);
            ah[mi] = u.v;
            u.q[0] = *(const uint4*)rl; u.q[1] = *(const uint4*)(rl + 16);
            al[mi] = u.v;
        }
#pragma unroll
        for (int ni = 0; ni < 4; ++ni) {
            const __bf16* rh = &Bh[bRow + n0 + ni * 16 + fm][kb];
            const __bf16* rl = &Bl[bRow + n0 + ni * 16 + fm][kb];
            union { v16bf v; uint4 q[2]; } u;
            u.q[0] = *(const uint4*)rh; u.q[1] = *(const uint4*)(rh + 16);
            bh[ni] = u.v;
            u.q[0] = *(const uint4*)rl; u.q[1] = *(const uint4*)(rl + 16);
            bl[ni] = u.v;
        }

        // ---- 24 WMMAs per wave per K-chunk ------------------------------
#pragma unroll
        for (int mi = 0; mi < 2; ++mi)
#pragma unroll
            for (int ni = 0; ni < 4; ++ni) {
                acc[mi][ni] = __builtin_amdgcn_wmma_f32_16x16x32_bf16(
                    false, ah[mi], false, bh[ni], (short)0, acc[mi][ni], false, false);
                acc[mi][ni] = __builtin_amdgcn_wmma_f32_16x16x32_bf16(
                    false, ah[mi], false, bl[ni], (short)0, acc[mi][ni], false, false);
                acc[mi][ni] = __builtin_amdgcn_wmma_f32_16x16x32_bf16(
                    false, al[mi], false, bh[ni], (short)0, acc[mi][ni], false, false);
            }
        __syncthreads();
        cur ^= 1;
    }

    // ---- epilogue: C layout VGPR r -> M = r (lanes 0-15) / r+8 (16-31) ----
    const int rowHalf = (lane >> 4) * 8;
    const int colLane = lane & 15;
#pragma unroll
    for (int ni = 0; ni < 4; ++ni) {
        const int col = n0 + ni * 16 + colLane;
        const float bv = bias[col];
#pragma unroll
        for (int mi = 0; mi < 2; ++mi)
#pragma unroll
            for (int r = 0; r < 8; ++r) {
                const int row = mBase + wm * 32 + mi * 16 + rowHalf + r;
                z[(size_t)row * Ndim + col] = acc[mi][ni][r] + bv;
            }
    }
}

// ---------------------------------------------------------------------------
// Kernel 2: bidirectional EMA scan + residual.
//   out[l,b,z] = fwd(g1) + bwd(g2) + beta*z ; chunked 2-level scan:
//   32 chunks x 128 steps, carries combined through LDS (decay^128).
//   block: 1024 threads = (32 chunks) x (32 z columns), one b per block.
// ---------------------------------------------------------------------------
__global__ __launch_bounds__(1024)
void ema_scan(const float* __restrict__ zbuf,
              const float* __restrict__ alpha,
              const float* __restrict__ beta,
              float*       __restrict__ out)
{
    constexpr int CH = 32;    // chunks
    constexpr int CL = 128;   // chunk length (CH * CL == L_DIM)

    __shared__ float cf[CH][33];
    __shared__ float cb[CH][33];

    const int tid   = threadIdx.x;
    const int ziL   = tid & 31;
    const int chunk = tid >> 5;
    const int b     = blockIdx.x >> 3;
    const int zg    = blockIdx.x & 7;
    const int zi    = zg * 32 + ziL;

    const float g1  = 1.0f / (1.0f + __expf(-alpha[zi]));
    const float g2  = 1.0f / (1.0f + __expf(-alpha[Z_DIM + zi]));
    const float d1  = 1.0f - g1;
    const float d2  = 1.0f - g2;
    const float bet = beta[zi];

    const int    l0      = chunk * CL;
    const size_t stride  = (size_t)B_DIM * Z_DIM;             // 2048 floats per l
    const size_t colBase = (size_t)b * Z_DIM + zi;

    // ---- pass 1: zero-initialized local scans, record chunk carries -------
    float h = 0.0f;
    {
        const float* p = zbuf + (size_t)l0 * stride + colBase;
        for (int j = 0; j < CL; ++j)
            h = fmaf(d1, h, g1 * p[(size_t)j * stride]);
    }
    cf[chunk][ziL] = h;

    float h2 = 0.0f;
    {
        const float* p = zbuf + (size_t)(l0 + CL - 1) * stride + colBase;
        for (int j = 0; j < CL; ++j)
            h2 = fmaf(d2, h2, g2 * p[-(ptrdiff_t)j * (ptrdiff_t)stride]);
    }
    cb[chunk][ziL] = h2;
    __syncthreads();

    // ---- pass 2: sequential carry scan across 32 chunks (per z column) ----
    if (tid < 32) {
        const float D1 = __powf(d1, (float)CL);
        float hin = 0.0f;
        for (int c = 0; c < CH; ++c) {
            const float t = cf[c][tid];
            cf[c][tid] = hin;                    // carry-in at chunk start
            hin = fmaf(hin, D1, t);
        }
    } else if (tid < 64) {
        const int lc = tid - 32;
        const float D2 = __powf(d2, (float)CL);
        float hin = 0.0f;
        for (int c = CH - 1; c >= 0; --c) {
            const float t = cb[c][lc];
            cb[c][lc] = hin;                     // carry-in from chunk end side
            hin = fmaf(hin, D2, t);
        }
    }
    __syncthreads();

    // ---- pass 3: re-run with carries; fuse residual and bwd add -----------
    h = cf[chunk][ziL];
    {
        const float* p = zbuf + (size_t)l0 * stride + colBase;
        float*       o = out  + (size_t)l0 * stride + colBase;
        for (int j = 0; j < CL; ++j) {
            const float zv = p[(size_t)j * stride];
            h = fmaf(d1, h, g1 * zv);
            o[(size_t)j * stride] = fmaf(bet, zv, h);
        }
    }
    h2 = cb[chunk][ziL];
    {
        const float* p = zbuf + (size_t)(l0 + CL - 1) * stride + colBase;
        float*       o = out  + (size_t)(l0 + CL - 1) * stride + colBase;
        for (int j = 0; j < CL; ++j) {
            const ptrdiff_t off = -(ptrdiff_t)j * (ptrdiff_t)stride;
            const float zv = p[off];
            h2 = fmaf(d2, h2, g2 * zv);
            o[off] += h2;                        // same-thread RMW, same-wave ordered
        }
    }
}

// ---------------------------------------------------------------------------
// launcher
// ---------------------------------------------------------------------------
extern "C" void kernel_launch(void* const* d_in, const int* in_sizes, int n_in,
                              void* d_out, int out_size, void* d_ws, size_t ws_size,
                              hipStream_t stream)
{
    (void)in_sizes; (void)n_in; (void)out_size; (void)ws_size;

    const float* x     = (const float*)d_in[0];   // (L, B, E)
    const float* alpha = (const float*)d_in[1];   // (2Z,)
    const float* beta  = (const float*)d_in[2];   // (Z,)
    const float* W     = (const float*)d_in[3];   // (Z, E)
    const float* bias  = (const float*)d_in[4];   // (Z,)
    float*       out   = (float*)d_out;           // (L, B, Z)

    char* ws = (char*)d_ws;
    float*  zbuf = (float*)ws;                                        // 32 MB
    const size_t zbytes = (size_t)M_DIM * Z_DIM * sizeof(float);
    __bf16* Whi = (__bf16*)(ws + zbytes);                             // 512 KB
    __bf16* Wlo = (__bf16*)(ws + zbytes + (size_t)Z_DIM * E_DIM * 2); // 512 KB

    const size_t gemmLds = 92160;   // A(2x5120) + Bh(2x20480) + Bl(2x20480)

    ema_convW<<<256, 256, 0, stream>>>(W, Whi, Wlo);
    ema_gemm <<<M_DIM / 64, 256, gemmLds, stream>>>(x, Whi, Wlo, bias, zbuf);
    ema_scan <<<B_DIM * (Z_DIM / 32), 1024, 0, stream>>>(zbuf, alpha, beta, out);
}
